// EquiSE3Transformer_2259152797793
// MI455X (gfx1250) — matrix operand order repaired and checked by reading, not compiled, not run on hardware
//
#include <hip/hip_runtime.h>
#include <hip/hip_fp16.h>

// ---------------------------------------------------------------------------
// SE(3)-Transformer forward for MI455X (gfx1250, wave32, WMMA).
// Radial MLPs (the dominant FLOPs) run on v_wmma_f32_16x16x32_f16, 16 edges
// per wave; packed MLP weights are staged block-wide into LDS with CDNA5
// global_load_async_to_lds_b128 (ASYNCcnt) so B operands come from LDS.
// Everything else (CG einsum, segment softmax, scatters) is VALU + atomics,
// which is correct for a memory-bound (~300MB basis stream) problem.
// The conv do=0 branch is dead code (output reads only h[1]) and is skipped.
// ---------------------------------------------------------------------------

typedef __attribute__((ext_vector_type(16))) _Float16 v16h;
typedef __attribute__((ext_vector_type(8)))  float    v8f;

#define DEV __device__ __forceinline__

constexpr int NN   = 8192;      // nodes
constexpr int EE   = 65536;     // edges
constexpr int FIN  = 32;        // degree-0 input channels
constexpr int CC   = 8;         // channels per degree
constexpr int CMK  = 2;         // attention key/value channels
constexpr int OUTC = 32;        // conv out channels
constexpr int ETILES = EE / 16; // 16 edges per WMMA tile
constexpr int MLP_WAVES = 4;    // waves per block in the WMMA MLP kernel

// ---- WMMA 16x16x32 f16 k-order maps used here (contiguous k = 16*g + t, A and B alike) --------
// A matrix (16x32, M=edges): lane = {g,m}; VGPR v holds K = base(v,g)+bit.
DEV int kmapA(int g, int t) {
  int v = t >> 1;
  return (g << 4) + t + 0 * v;
}
// B matrix (32x16, N=outputs): lane = column; lanes 0-15 K=0..15, 16-31 K=16..31.
DEV int kmapB(int g, int t) { return (g << 4) + t; }

// ---- monotonic float<->uint encoding for atomicMax-based segment max ------
DEV unsigned fenc(float f) {
  unsigned u = __float_as_uint(f);
  return (u & 0x80000000u) ? ~u : (u | 0x80000000u);
}
DEV float fdec(unsigned u) {
  return __uint_as_float((u & 0x80000000u) ? (u & 0x7FFFFFFFu) : ~u);
}

// ===========================================================================
// Packing kernels
// ===========================================================================

// edge_in = [r, edge_w] padded to K=32, stored per 16-edge tile in WMMA-A
// lane layout so the MLP kernel does a single v16h load per wave-lane.
__global__ void pack_edge_in_kernel(const float* __restrict__ r,
                                    const float* __restrict__ ew,
                                    _Float16* __restrict__ A) {
  int idx = blockIdx.x * blockDim.x + threadIdx.x;    // ETILES*32
  if (idx >= ETILES * 32) return;
  int lane = idx & 31, g = lane >> 4, m = lane & 15;
  int tile = idx >> 5;
  int e = tile * 16 + m;
  v16h v;
#pragma unroll
  for (int t = 0; t < 16; t++) {
    int K = kmapA(g, t);
    float x = (K == 0) ? r[e] : (K <= 4 ? ew[e * 4 + (K - 1)] : 0.f);
    v[t] = (_Float16)x;
  }
  *reinterpret_cast<v16h*>(A + (size_t)idx * 16) = v;
}

// Pack one radial MLP (W1[5,32], W2[32,32], W3[32,outN]) into f16 B tiles.
// Tile order: [0,1]=W1, [2,3]=W2, [4..]=W3.
__global__ void pack_mlp_kernel(const float* __restrict__ W1,
                                const float* __restrict__ W2,
                                const float* __restrict__ W3,
                                _Float16* __restrict__ out, int outN) {
  int idx = blockIdx.x * blockDim.x + threadIdx.x;
  int totTiles = 4 + outN / 16;
  if (idx >= totTiles * 32) return;
  int lane = idx & 31, g = lane >> 4, n = lane & 15;
  int nt = idx >> 5;
  const float* W; int Krows, wN, col;
  if (nt < 2)      { W = W1; Krows = 5;  wN = 32;   col = nt * 16 + n; }
  else if (nt < 4) { W = W2; Krows = 32; wN = 32;   col = (nt - 2) * 16 + n; }
  else             { W = W3; Krows = 32; wN = outN; col = (nt - 4) * 16 + n; }
  v16h v;
#pragma unroll
  for (int t = 0; t < 16; t++) {
    int K = kmapB(g, t);
    v[t] = (K < Krows) ? (_Float16)W[K * wN + col] : (_Float16)0.f;
  }
  *reinterpret_cast<v16h*>(out + (size_t)idx * 16) = v;
}

// ===========================================================================
// Radial-MLP WMMA kernel: 16 edges/wave, 5->32->32->outN, relu on hiddens.
// Packed weights are staged block-wide into LDS via CDNA5 async-to-LDS DMA
// (tracked by ASYNCcnt), then consumed as WMMA B operands from LDS.
// Output w stored f16 as [E][outN] (outN = nf*co*ci).
// ===========================================================================
__global__ void __launch_bounds__(MLP_WAVES * 32)
mlp_wmma_kernel(const _Float16* __restrict__ Ain,
                const _Float16* __restrict__ Wp,
                const float* __restrict__ b1,
                const float* __restrict__ b2,
                const float* __restrict__ b3,
                _Float16* __restrict__ Wout, int outTiles) {
  extern __shared__ _Float16 wlds[];                 // (4+outTiles)*512 halfs
  __shared__ _Float16 st1[MLP_WAVES][16][32];
  __shared__ _Float16 st2[MLP_WAVES][16][32];
  int lane = threadIdx.x & 31, wave = threadIdx.x >> 5;
  int tile = blockIdx.x * MLP_WAVES + wave;          // grid is exact: ETILES/MLP_WAVES
  int m = lane & 15, g = lane >> 4;

  // ---- stage packed weights into LDS with async DMA (gfx1250 path) -------
  {
    int chunks = (4 + outTiles) * 64;                // 16B chunks
    unsigned lbase = (unsigned)(size_t)(void*)wlds;  // generic LDS addr low bits
    unsigned long long gbase = (unsigned long long)(const void*)Wp;
    for (int c0 = (int)threadIdx.x; c0 < chunks; c0 += (int)blockDim.x) {
      unsigned l = lbase + (unsigned)c0 * 16u;
      unsigned long long ga = gbase + (unsigned long long)c0 * 16ull;
      asm volatile("global_load_async_to_lds_b128 %0, %1, off"
                   :: "v"(l), "v"(ga) : "memory");
    }
    asm volatile("s_wait_asynccnt 0" ::: "memory");
  }
  __syncthreads();

  const v16h* Av = reinterpret_cast<const v16h*>(Ain);
  const v16h* Wv = reinterpret_cast<const v16h*>(wlds);   // B tiles from LDS

  v16h a0 = Av[(size_t)tile * 32 + lane];

  { // layer 1: 32 -> 32 (two N tiles), bias + relu, restage f16
    v8f z = {};
    v8f c0 = __builtin_amdgcn_wmma_f32_16x16x32_f16(false, a0, false, Wv[0 * 32 + lane], (short)0, z, false, false);
    v8f c1 = __builtin_amdgcn_wmma_f32_16x16x32_f16(false, a0, false, Wv[1 * 32 + lane], (short)0, z, false, false);
    float bb0 = b1[m], bb1 = b1[16 + m];
#pragma unroll
    for (int rI = 0; rI < 8; rI++) {
      int row = rI + g * 8;
      st1[wave][row][m]      = (_Float16)fmaxf(c0[rI] + bb0, 0.f);
      st1[wave][row][16 + m] = (_Float16)fmaxf(c1[rI] + bb1, 0.f);
    }
  }
  __syncthreads();
  v16h a1;
#pragma unroll
  for (int t = 0; t < 16; t++) a1[t] = st1[wave][m][kmapA(g, t)];

  { // layer 2
    v8f z = {};
    v8f c0 = __builtin_amdgcn_wmma_f32_16x16x32_f16(false, a1, false, Wv[2 * 32 + lane], (short)0, z, false, false);
    v8f c1 = __builtin_amdgcn_wmma_f32_16x16x32_f16(false, a1, false, Wv[3 * 32 + lane], (short)0, z, false, false);
    float bb0 = b2[m], bb1 = b2[16 + m];
#pragma unroll
    for (int rI = 0; rI < 8; rI++) {
      int row = rI + g * 8;
      st2[wave][row][m]      = (_Float16)fmaxf(c0[rI] + bb0, 0.f);
      st2[wave][row][16 + m] = (_Float16)fmaxf(c1[rI] + bb1, 0.f);
    }
  }
  __syncthreads();
  v16h a2;
#pragma unroll
  for (int t = 0; t < 16; t++) a2[t] = st2[wave][m][kmapA(g, t)];

  // layer 3: outTiles WMMAs, bias, store f16
  int outN = outTiles * 16;
  for (int nt = 0; nt < outTiles; nt++) {
    v8f z = {};
    v8f c = __builtin_amdgcn_wmma_f32_16x16x32_f16(false, a2, false, Wv[(4 + nt) * 32 + lane], (short)0, z, false, false);
    float bb = b3[nt * 16 + m];
#pragma unroll
    for (int rI = 0; rI < 8; rI++) {
      int e = tile * 16 + rI + g * 8;
      Wout[(size_t)e * outN + nt * 16 + m] = (_Float16)(c[rI] + bb);
    }
  }
}

// ===========================================================================
// CG einsum per edge: msg[e,c,o] += sum_{f,i,d} basis[e,f,o,i]*w[e,f,c,d]*fs[e,d,i]
// fs = h[di][src[e]].  Fully templated so everything stays in registers.
// w is single-use (just produced, L2-resident) -> non-temporal loads.
// ===========================================================================
template <int NF, int O, int I, int CI, int CO>
__global__ void einsum_kernel(const _Float16* __restrict__ w,
                              const float* __restrict__ basis,
                              const float* __restrict__ hsrc,
                              const int* __restrict__ src,
                              float* __restrict__ msg) {
  int e = blockIdx.x * blockDim.x + threadIdx.x;
  if (e >= EE) return;
  int s = src[e];
  const float* hp = hsrc + (size_t)s * CI * I;
  float fs[CI * I];
#pragma unroll
  for (int j = 0; j < CI * I; j++) fs[j] = hp[j];

  float* mp = msg + (size_t)e * CO * O;
  float acc[CO * O];
#pragma unroll
  for (int j = 0; j < CO * O; j++) acc[j] = mp[j];

  const _Float16* wp = w + (size_t)e * (NF * CO * CI);
  const float* bp = basis + (size_t)e * (NF * O * I);
  __builtin_prefetch(bp + (size_t)256 * NF * O * I, 0, 0);  // global_prefetch_b8

#pragma unroll
  for (int f = 0; f < NF; f++) {
    float bs[O * I];
#pragma unroll
    for (int j = 0; j < O * I; j++) bs[j] = bp[f * O * I + j];
#pragma unroll
    for (int d = 0; d < CI; d++) {
      float gv[O];
#pragma unroll
      for (int o = 0; o < O; o++) {
        float sacc = 0.f;
#pragma unroll
        for (int i = 0; i < I; i++) sacc += bs[o * I + i] * fs[d * I + i];
        gv[o] = sacc;
      }
#pragma unroll
      for (int c = 0; c < CO; c++) {
        float wv = (float)__builtin_nontemporal_load(&wp[(f * CO + c) * CI + d]);
#pragma unroll
        for (int o = 0; o < O; o++) acc[c * O + o] += wv * gv[o];
      }
    }
  }
#pragma unroll
  for (int j = 0; j < CO * O; j++) mp[j] = acc[j];
}

// ===========================================================================
// Small VALU kernels: fill, q-projection, attention, scatters, norms, etc.
// ===========================================================================
__global__ void fill_u32_kernel(unsigned* __restrict__ p, unsigned v, long long n) {
  long long i = blockIdx.x * (long long)blockDim.x + threadIdx.x;
  if (i < n) p[i] = v;
}

// q[n,o,m] = sum_c Wq[o,c] * h[n,c,m]
__global__ void qproj_kernel(const float* __restrict__ Wq, const float* __restrict__ h,
                             float* __restrict__ q, int ci, int M) {
  int idx = blockIdx.x * blockDim.x + threadIdx.x;  // NN*CMK*M
  if (idx >= NN * CMK * M) return;
  int mm = idx % M; int t = idx / M; int o = t % CMK; int n = t / CMK;
  float s = 0.f;
  for (int c = 0; c < ci; c++) s += Wq[o * ci + c] * h[((size_t)n * ci + c) * M + mm];
  q[idx] = s;
}

struct P4 { const float* p[4]; };

__global__ void attn_dot_kernel(P4 kb, P4 qb, const int* __restrict__ dst,
                                float* __restrict__ er, int ndeg, float scale) {
  int e = blockIdx.x * blockDim.x + threadIdx.x;
  if (e >= EE) return;
  int dn = dst[e];
  float s = 0.f;
  for (int d = 0; d < ndeg; d++) {
    int M = 2 * d + 1;
    const float* kk = kb.p[d] + (size_t)e * CMK * M;
    const float* qq = qb.p[d] + (size_t)dn * CMK * M;
    for (int j = 0; j < CMK * M; j++) s += kk[j] * qq[j];
  }
  er[e] = s * scale;
}

__global__ void attn_max_kernel(const float* __restrict__ er, const int* __restrict__ dst,
                                unsigned* __restrict__ menc) {
  int e = blockIdx.x * blockDim.x + threadIdx.x;
  if (e >= EE) return;
  atomicMax(&menc[dst[e]], fenc(er[e]));
}

__global__ void attn_expsum_kernel(const float* __restrict__ er, const int* __restrict__ dst,
                                   const unsigned* __restrict__ menc,
                                   float* __restrict__ ex, float* __restrict__ ssum) {
  int e = blockIdx.x * blockDim.x + threadIdx.x;
  if (e >= EE) return;
  int dn = dst[e];
  float v = __expf(er[e] - fdec(menc[dn]));
  ex[e] = v;
  atomicAdd(&ssum[dn], v);
}

__global__ void attn_norm_kernel(float* __restrict__ ex, const int* __restrict__ dst,
                                 const float* __restrict__ ssum) {
  int e = blockIdx.x * blockDim.x + threadIdx.x;
  if (e >= EE) return;
  ex[e] = ex[e] / (ssum[dst[e]] + 1e-9f);
}

// z[dst[e]] += coef[e] * v[e]  (coef==nullptr -> plain segment sum)
__global__ void seg_scatter_kernel(const float* __restrict__ coef, const float* __restrict__ v,
                                   const int* __restrict__ dst, float* __restrict__ z, int S) {
  long long idx = blockIdx.x * (long long)blockDim.x + threadIdx.x;  // EE*S
  if (idx >= (long long)EE * S) return;
  int e = (int)(idx / S); int j = (int)(idx % S);
  float val = v[idx];
  if (coef) val *= coef[e];
  atomicAdd(&z[(size_t)dst[e] * S + j], val);
}

__global__ void count_deg_kernel(const int* __restrict__ dst, float* __restrict__ deg) {
  int e = blockIdx.x * blockDim.x + threadIdx.x;
  if (e >= EE) return;
  atomicAdd(&deg[dst[e]], 1.f);
}

// out[n,o,m] = sum_{c<CMK} Wp[o,c]*z[n,c,m] + sum_{c<cs} Wp[o,CMK+c]*hs[n,c,m]
__global__ void proj_cat_kernel(const float* __restrict__ Wp, const float* __restrict__ z,
                                const float* __restrict__ hs, float* __restrict__ out,
                                int cs, int M) {
  int idx = blockIdx.x * blockDim.x + threadIdx.x;  // NN*CC*M
  if (idx >= NN * CC * M) return;
  int mm = idx % M; int t = idx / M; int o = t % CC; int n = t / CC;
  int wi = CMK + cs;
  float s = 0.f;
  for (int c = 0; c < CMK; c++) s += Wp[o * wi + c] * z[((size_t)n * CMK + c) * M + mm];
  if (hs)
    for (int c = 0; c < cs; c++) s += Wp[o * wi + CMK + c] * hs[((size_t)n * cs + c) * M + mm];
  out[idx] = s;
}

// norm-based equivariant nonlinearity
__global__ void gnorm_kernel(const float* __restrict__ g, const float* __restrict__ b,
                             const float* __restrict__ W, const float* __restrict__ bW,
                             const float* __restrict__ x, float* __restrict__ out, int M) {
  int n = blockIdx.x * blockDim.x + threadIdx.x;
  if (n >= NN) return;
  const float* xp = x + (size_t)n * CC * M;
  float* op = out + (size_t)n * CC * M;
  float nrm[CC], ln[CC];
  float mu = 0.f;
  for (int c = 0; c < CC; c++) {
    float s2 = 1e-12f;
    for (int m = 0; m < M; m++) { float v = xp[c * M + m]; s2 += v * v; }
    nrm[c] = sqrtf(s2);
    mu += nrm[c];
  }
  mu *= (1.f / CC);
  float var = 0.f;
  for (int c = 0; c < CC; c++) { float d = nrm[c] - mu; var += d * d; }
  var *= (1.f / CC);
  float inv = rsqrtf(var + 1e-5f);
  for (int c = 0; c < CC; c++) ln[c] = (nrm[c] - mu) * inv * g[c] + b[c];
  for (int o = 0; o < CC; o++) {
    float s = bW[o];
    for (int c = 0; c < CC; c++) s += ln[c] * W[c * CC + o];
    s = fmaxf(s, 0.f);
    float sc = s / nrm[o];
    for (int m = 0; m < M; m++) op[o * M + m] = xp[o * M + m] * sc;
  }
}

// out[n,o,m] = acc[n,o,m]/max(deg,1) + sum_c self[o,c]*h[n,c,m]
__global__ void conv_finish_kernel(const float* __restrict__ acc, const float* __restrict__ deg,
                                   const float* __restrict__ self, const float* __restrict__ h,
                                   float* __restrict__ out, int M) {
  int idx = blockIdx.x * blockDim.x + threadIdx.x;  // NN*OUTC*M
  if (idx >= NN * OUTC * M) return;
  int mm = idx % M; int t = idx / M; int o = t % OUTC; int n = t / OUTC;
  float s = acc[idx] / fmaxf(deg[n], 1.f);
  for (int c = 0; c < CC; c++) s += self[o * CC + c] * h[((size_t)n * CC + c) * M + mm];
  out[idx] = s;
}

// final_pos[n,m] = sum_c out_w[c] * h1[n,c,m]  (M=3)
__global__ void final_kernel(const float* __restrict__ ow, const float* __restrict__ h1,
                             float* __restrict__ out) {
  int idx = blockIdx.x * blockDim.x + threadIdx.x;  // NN*3
  if (idx >= NN * 3) return;
  int mm = idx % 3, n = idx / 3;
  float s = 0.f;
#pragma unroll
  for (int c = 0; c < OUTC; c++) s += ow[c] * h1[(size_t)n * OUTC * 3 + c * 3 + mm];
  out[idx] = s;
}

// ===========================================================================
// Host orchestration
// ===========================================================================
static inline int gsz(long long n, int b) { return (int)((n + b - 1) / b); }

static void launch_einsum(int nf, int O, int I, int ci, int co,
                          const _Float16* w, const float* basis, const float* hsrc,
                          const int* src, float* msg, hipStream_t s) {
  dim3 blk(256), grd(gsz(EE, 256));
#define EINS(a1, a2, a3, a4, a5)                                               \
  if (nf == a1 && O == a2 && I == a3 && ci == a4 && co == a5) {                \
    einsum_kernel<a1, a2, a3, a4, a5><<<grd, blk, 0, s>>>(w, basis, hsrc, src, msg); \
    return;                                                                    \
  }
  // l1 (ci=32, co=2)
  EINS(1, 1, 1, 32, 2) EINS(1, 3, 1, 32, 2) EINS(1, 5, 1, 32, 2) EINS(1, 7, 1, 32, 2)
  // l2 (ci=8, co=2)
  EINS(1, 1, 1, 8, 2) EINS(1, 3, 1, 8, 2) EINS(1, 5, 1, 8, 2) EINS(1, 7, 1, 8, 2)
  EINS(1, 1, 3, 8, 2) EINS(3, 3, 3, 8, 2) EINS(3, 5, 3, 8, 2) EINS(3, 7, 3, 8, 2)
  EINS(1, 1, 5, 8, 2) EINS(3, 3, 5, 8, 2) EINS(5, 5, 5, 8, 2) EINS(5, 7, 5, 8, 2)
  EINS(1, 1, 7, 8, 2) EINS(3, 3, 7, 8, 2) EINS(5, 5, 7, 8, 2) EINS(7, 7, 7, 8, 2)
  // conv do=1 (ci=8, co=32)
  EINS(1, 3, 1, 8, 32) EINS(3, 3, 3, 8, 32) EINS(3, 3, 5, 8, 32) EINS(3, 3, 7, 8, 32)
#undef EINS
}

// pack weights -> WMMA MLP -> CG einsum (accumulating into msg)
static void run_pair(void* const* din, int base, int outN,
                     int nf, int O, int I, int ci, int co,
                     const float* basis, const float* hsrc, const int* src,
                     float* msg, const _Float16* Aedge, _Float16*& wslot,
                     _Float16* wbuf, hipStream_t st) {
  const float* W1 = (const float*)din[base + 0];
  const float* b1 = (const float*)din[base + 1];
  const float* W2 = (const float*)din[base + 2];
  const float* b2 = (const float*)din[base + 3];
  const float* W3 = (const float*)din[base + 4];
  const float* b3 = (const float*)din[base + 5];
  int outTiles = outN / 16;
  int totTiles = 4 + outTiles;
  pack_mlp_kernel<<<gsz(totTiles * 32, 128), 128, 0, st>>>(W1, W2, W3, wslot, outN);
  size_t shbytes = (size_t)totTiles * 512 * sizeof(_Float16);  // weight LDS stage
  mlp_wmma_kernel<<<ETILES / MLP_WAVES, MLP_WAVES * 32, shbytes, st>>>(
      Aedge, wslot, b1, b2, b3, wbuf, outTiles);
  launch_einsum(nf, O, I, ci, co, wbuf, basis, hsrc, src, msg, st);
  wslot += (size_t)totTiles * 512;  // 32 lanes * 16 halfs per tile
}

extern "C" void kernel_launch(void* const* d_in, const int* in_sizes, int n_in,
                              void* d_out, int out_size, void* d_ws, size_t ws_size,
                              hipStream_t stream) {
  (void)in_sizes; (void)n_in; (void)out_size; (void)ws_size;

  // ---- input leaf indices (pytree depth-first, dict insertion order) ------
  const float* feat   = (const float*)d_in[0];   // [N,32,1]
  const float* edge_w = (const float*)d_in[1];   // [E,4]
  const float* r      = (const float*)d_in[2];   // [E,1]
  const int*   src    = (const int*)d_in[3];
  const int*   dst    = (const int*)d_in[4];
  auto basis = [&](int di, int dO) { return (const float*)d_in[5 + di * 4 + dO]; };
  // l1: q=21, k(0,0)=22..27, v(0,d)=28+6d, proj(d)=52+d; n1(d)=56+4d
  // l2: q(d)=72+d, k(di,do)=76+(di*4+do)*6, v=172+..., proj(d)=268+d, n2(d)=272+4d
  // conv: pair(di,do)=288+(di*2+do)*6, self1=337, out_w=338

  // ---- workspace bump allocator -------------------------------------------
  char* wsb = (char*)d_ws; size_t off = 0;
  auto alloc = [&](size_t bytes) -> char* {
    off = (off + 255) & ~(size_t)255; char* p = wsb + off; off += bytes; return p;
  };
  _Float16* Aedge  = (_Float16*)alloc((size_t)ETILES * 32 * 16 * 2);  // 4MB
  _Float16* warena = (_Float16*)alloc((size_t)1 << 20);               // packed weights
  _Float16* wbuf   = (_Float16*)alloc((size_t)EE * 768 * 2);          // MLP out (f16)
  float *hA[4], *hB[4], *hC[4], *hD[4], *kb[4], *vb[4], *qb[4], *zb[4];
  for (int d = 0; d < 4; d++) {
    int M = 2 * d + 1;
    hA[d] = (float*)alloc((size_t)NN * CC * M * 4);
    hB[d] = (float*)alloc((size_t)NN * CC * M * 4);
    hC[d] = (float*)alloc((size_t)NN * CC * M * 4);
    hD[d] = (float*)alloc((size_t)NN * CC * M * 4);
    kb[d] = (float*)alloc((size_t)EE * CMK * M * 4);
    vb[d] = (float*)alloc((size_t)EE * CMK * M * 4);
    qb[d] = (float*)alloc((size_t)NN * CMK * M * 4);
    zb[d] = (float*)alloc((size_t)NN * CMK * M * 4);
  }
  float* hE1  = (float*)alloc((size_t)NN * OUTC * 3 * 4);
  float* er   = (float*)alloc((size_t)EE * 4);
  float* ab   = (float*)alloc((size_t)EE * 4);
  unsigned* menc = (unsigned*)alloc((size_t)NN * 4);
  float* ssum = (float*)alloc((size_t)NN * 4);
  float* deg  = (float*)alloc((size_t)NN * 4);
  float* acc1 = (float*)alloc((size_t)NN * OUTC * 3 * 4);
  float* mb1  = (float*)alloc((size_t)EE * OUTC * 3 * 4);

  auto zerou = [&](void* p, long long n) {
    fill_u32_kernel<<<gsz(n, 256), 256, 0, stream>>>((unsigned*)p, 0u, n);
  };

  _Float16* wslot = warena;
  pack_edge_in_kernel<<<gsz(ETILES * 32, 256), 256, 0, stream>>>(r, edge_w, Aedge);

  // ======================= Layer 1 (dins=[0]) ==============================
  zerou(kb[0], (long long)EE * CMK * 1);
  for (int d = 0; d < 4; d++) zerou(vb[d], (long long)EE * CMK * (2 * d + 1));
  run_pair(d_in, 22, 64, 1, 1, 1, 32, 2, basis(0, 0), feat, src, kb[0], Aedge, wslot, wbuf, stream);
  for (int d = 0; d < 4; d++)
    run_pair(d_in, 28 + 6 * d, 64, 1, 2 * d + 1, 1, 32, 2, basis(0, d), feat, src, vb[d],
             Aedge, wslot, wbuf, stream);
  qproj_kernel<<<gsz(NN * CMK * 1, 256), 256, 0, stream>>>((const float*)d_in[21], feat, qb[0], 32, 1);
  {
    P4 kk{{kb[0], nullptr, nullptr, nullptr}};
    P4 qq{{qb[0], nullptr, nullptr, nullptr}};
    attn_dot_kernel<<<gsz(EE, 256), 256, 0, stream>>>(kk, qq, dst, er, 1, 1.f / sqrtf(2.f));
  }
  zerou(menc, NN); zerou(ssum, NN);
  attn_max_kernel<<<gsz(EE, 256), 256, 0, stream>>>(er, dst, menc);
  attn_expsum_kernel<<<gsz(EE, 256), 256, 0, stream>>>(er, dst, menc, ab, ssum);
  attn_norm_kernel<<<gsz(EE, 256), 256, 0, stream>>>(ab, dst, ssum);
  for (int d = 0; d < 4; d++) {
    int S = CMK * (2 * d + 1);
    zerou(zb[d], (long long)NN * S);
    seg_scatter_kernel<<<gsz((long long)EE * S, 256), 256, 0, stream>>>(ab, vb[d], dst, zb[d], S);
  }
  proj_cat_kernel<<<gsz(NN * CC * 1, 256), 256, 0, stream>>>((const float*)d_in[52], zb[0], feat, hA[0], 32, 1);
  for (int d = 1; d < 4; d++)
    proj_cat_kernel<<<gsz(NN * CC * (2 * d + 1), 256), 256, 0, stream>>>(
        (const float*)d_in[52 + d], zb[d], nullptr, hA[d], 0, 2 * d + 1);
  for (int d = 0; d < 4; d++)
    gnorm_kernel<<<gsz(NN, 128), 128, 0, stream>>>(
        (const float*)d_in[56 + 4 * d], (const float*)d_in[57 + 4 * d],
        (const float*)d_in[58 + 4 * d], (const float*)d_in[59 + 4 * d], hA[d], hB[d], 2 * d + 1);

  // ======================= Layer 2 (all degrees) ===========================
  for (int d = 0; d < 4; d++) {
    zerou(kb[d], (long long)EE * CMK * (2 * d + 1));
    zerou(vb[d], (long long)EE * CMK * (2 * d + 1));
  }
  for (int dO = 0; dO < 4; dO++)
    for (int di = 0; di < 4; di++) {
      int mn = di < dO ? di : dO;
      int nf = 2 * mn + 1, O = 2 * dO + 1, I = 2 * di + 1;
      run_pair(d_in, 76 + (di * 4 + dO) * 6, nf * 16, nf, O, I, 8, 2,
               basis(di, dO), hB[di], src, kb[dO], Aedge, wslot, wbuf, stream);
      run_pair(d_in, 172 + (di * 4 + dO) * 6, nf * 16, nf, O, I, 8, 2,
               basis(di, dO), hB[di], src, vb[dO], Aedge, wslot, wbuf, stream);
    }
  for (int d = 0; d < 4; d++)
    qproj_kernel<<<gsz(NN * CMK * (2 * d + 1), 256), 256, 0, stream>>>(
        (const float*)d_in[72 + d], hB[d], qb[d], 8, 2 * d + 1);
  {
    P4 kk{{kb[0], kb[1], kb[2], kb[3]}};
    P4 qq{{qb[0], qb[1], qb[2], qb[3]}};
    attn_dot_kernel<<<gsz(EE, 256), 256, 0, stream>>>(kk, qq, dst, er, 4, 1.f / sqrtf(32.f));
  }
  zerou(menc, NN); zerou(ssum, NN);
  attn_max_kernel<<<gsz(EE, 256), 256, 0, stream>>>(er, dst, menc);
  attn_expsum_kernel<<<gsz(EE, 256), 256, 0, stream>>>(er, dst, menc, ab, ssum);
  attn_norm_kernel<<<gsz(EE, 256), 256, 0, stream>>>(ab, dst, ssum);
  for (int d = 0; d < 4; d++) {
    int S = CMK * (2 * d + 1);
    zerou(zb[d], (long long)NN * S);
    seg_scatter_kernel<<<gsz((long long)EE * S, 256), 256, 0, stream>>>(ab, vb[d], dst, zb[d], S);
  }
  for (int d = 0; d < 4; d++)
    proj_cat_kernel<<<gsz(NN * CC * (2 * d + 1), 256), 256, 0, stream>>>(
        (const float*)d_in[268 + d], zb[d], hB[d], hC[d], 8, 2 * d + 1);
  for (int d = 0; d < 4; d++)
    gnorm_kernel<<<gsz(NN, 128), 128, 0, stream>>>(
        (const float*)d_in[272 + 4 * d], (const float*)d_in[273 + 4 * d],
        (const float*)d_in[274 + 4 * d], (const float*)d_in[275 + 4 * d], hC[d], hD[d], 2 * d + 1);

  // ======================= Conv (only do=1 reaches the output) =============
  zerou(mb1, (long long)EE * OUTC * 3);
  for (int di = 0; di < 4; di++) {
    int mn = di < 1 ? di : 1;
    int nf = 2 * mn + 1;
    run_pair(d_in, 288 + (di * 2 + 1) * 6, nf * 256, nf, 3, 2 * di + 1, 8, 32,
             basis(di, 1), hD[di], src, mb1, Aedge, wslot, wbuf, stream);
  }
  zerou(deg, NN);
  count_deg_kernel<<<gsz(EE, 256), 256, 0, stream>>>(dst, deg);
  zerou(acc1, (long long)NN * OUTC * 3);
  seg_scatter_kernel<<<gsz((long long)EE * OUTC * 3, 256), 256, 0, stream>>>(
      nullptr, mb1, dst, acc1, OUTC * 3);
  conv_finish_kernel<<<gsz(NN * OUTC * 3, 256), 256, 0, stream>>>(
      acc1, deg, (const float*)d_in[337], hD[1], hE1, 3);

  final_kernel<<<gsz(NN * 3, 256), 256, 0, stream>>>((const float*)d_in[338], hE1, (float*)d_out);
}
